// LstmDecoder_76811195122550
// MI455X (gfx1250) — compile-verified, matrix operand order checked by reading, not run on hardware
//
#include <hip/hip_runtime.h>
#include <hip/hip_bf16.h>
#include <math.h>

// ---------------- problem constants ----------------
#define B_   128
#define T_   1024
#define H_   512
#define G4_  2048   // 4*H gates
#define KX_  256    // sequence-feature part of the LSTM input
#define OUT_ 256
#define ZD_  256
#define NWG_ 256    // 8 M-tiles x 32 H-slices

// ---------------- TDM availability (device pass only) ----------------
#ifndef __has_builtin
#define __has_builtin(x) 0
#endif
#if __has_builtin(__builtin_amdgcn_tensor_load_to_lds) && __has_builtin(__builtin_amdgcn_s_wait_tensorcnt)
#define HAVE_TDM 1
#else
#define HAVE_TDM 0
#endif

// ---------------- vector types (POD, union-safe) ----------------
typedef __attribute__((ext_vector_type(16))) __bf16       v16bf;
typedef __attribute__((ext_vector_type(8)))  float        v8f;
typedef __attribute__((ext_vector_type(4)))  unsigned int u32x4;
typedef __attribute__((ext_vector_type(8)))  int          i32x8;
typedef __attribute__((ext_vector_type(4)))  int          i32x4;
typedef __attribute__((ext_vector_type(4)))  float        f32x4;

union FragBF { v16bf v; u32x4 q[2]; };
union FragF  { v8f   v; f32x4 q[2]; };

// ---------------- workspace layout (bytes) ----------------
static constexpr size_t HS_OFF    = 0;
static constexpr size_t HS_BYTES  = (size_t)(T_ + 1) * B_ * H_ * 2;   // ~128 MB
static constexpr size_t C_OFF     = HS_OFF + HS_BYTES;                // c in fragment layout
static constexpr size_t C_BYTES   = (size_t)B_ * H_ * 4;
static constexpr size_t ZC_OFF    = C_OFF + C_BYTES;                  // Zc in fragment layout
static constexpr size_t ZC_BYTES  = (size_t)B_ * G4_ * 4;
static constexpr size_t WIH_OFF   = ZC_OFF + ZC_BYTES;
static constexpr size_t WIH_BYTES = (size_t)G4_ * KX_ * 2;
static constexpr size_t WHH_OFF   = WIH_OFF + WIH_BYTES;
static constexpr size_t WHH_BYTES = (size_t)G4_ * H_ * 2;
static constexpr size_t WOUT_OFF  = WHH_OFF + WHH_BYTES;
// total ~133 MiB

// ---------------- helpers ----------------
__device__ __forceinline__ float fsig(float x) { return 1.0f / (1.0f + __expf(-x)); }
__device__ __forceinline__ float ftanh(float x) {
    float e = __expf(-2.0f * fabsf(x));
    float t = (1.0f - e) / (1.0f + e);
    return copysignf(t, x);
}
__device__ __forceinline__ v8f wmma_bf16(v16bf a, v16bf b, v8f c) {
    return __builtin_amdgcn_wmma_f32_16x16x32_bf16(false, a, false, b, (short)0, c, false, false);
}

// Fragment-layout index for a (row, col) element of a [B_][H_] state tensor.
__device__ __host__ __forceinline__ size_t frag_idx_bh(int row, int col) {
    int wg   = (col >> 4) * 8 + (row >> 4);
    int lane = ((row & 15) >> 3) * 16 + (col & 15);
    int i    = row & 7;
    return ((size_t)(wg * 32 + lane) << 3) + i;
}

// ---------------- K1: h0/c0 from embedding (c written in fragment layout) ----------------
__global__ void embed_init(const float* __restrict__ z, const float* __restrict__ We,
                           const float* __restrict__ be,
                           __bf16* __restrict__ hs0, float* __restrict__ cfrag) {
    int tid = blockIdx.x * blockDim.x + threadIdx.x;
    if (tid >= B_ * H_) return;
    int b = tid / H_, j = tid % H_;
    const float* zr = z  + (size_t)b * ZD_;
    const float* w0 = We + (size_t)j * ZD_;
    const float* w1 = We + (size_t)(H_ + j) * ZD_;
    float a0 = be[j], a1 = be[H_ + j];
    for (int k = 0; k < ZD_; ++k) { float zv = zr[k]; a0 += zv * w0[k]; a1 += zv * w1[k]; }
    hs0[(size_t)b * H_ + j]  = (__bf16)ftanh(a0);
    cfrag[frag_idx_bh(b, j)] = ftanh(a1);
}

// ---------------- K2a: weight conversion to bf16 ----------------
__global__ void prep_weights(const float* __restrict__ Wih_f, const float* __restrict__ Whh_f,
                             const float* __restrict__ Wout_f,
                             __bf16* __restrict__ Wih_b, __bf16* __restrict__ Whh_b,
                             __bf16* __restrict__ Wout_b) {
    const int R1 = G4_ * KX_, R2 = G4_ * H_, R3 = OUT_ * H_;
    int tid = blockIdx.x * blockDim.x + threadIdx.x;
    if (tid < R1) {
        int col = tid / KX_, k = tid % KX_;                 // only the seq-feature half of W_ih
        Wih_b[tid] = (__bf16)Wih_f[(size_t)col * 512 + k];
    } else if (tid < R1 + R2) {
        int r = tid - R1;
        Whh_b[r] = (__bf16)Whh_f[r];
    } else if (tid < R1 + R2 + R3) {
        int r = tid - R1 - R2;
        Wout_b[r] = (__bf16)Wout_f[r];
    }
}

// ---- K2b: Zc = z @ W_ih[:,256:]^T + b_ih + b_hh, written in WMMA C-fragment layout ----
__global__ void zc_kernel(const float* __restrict__ z, const float* __restrict__ Wih_f,
                          const float* __restrict__ b_ih, const float* __restrict__ b_hh,
                          float* __restrict__ Zcf) {
    int tid = blockIdx.x * blockDim.x + threadIdx.x;
    if (tid >= B_ * G4_) return;
    int i    = tid & 7;
    int lane = (tid >> 3) & 31;
    int g    = (tid >> 8) & 3;
    int wg   = tid >> 10;
    int m0   = (wg & 7) * 16;
    int hc0  = (wg >> 3) * 16;
    int n    = lane & 15;
    int half = lane >> 4;
    int row  = m0 + i + 8 * half;
    int col  = g * H_ + hc0 + n;
    const float* zr = z + (size_t)row * ZD_;
    const float* wr = Wih_f + (size_t)col * 512 + 256;      // z part of the input
    float a = b_ih[col] + b_hh[col];
    for (int k = 0; k < ZD_; ++k) a += zr[k] * wr[k];
    Zcf[tid] = a;
}

// ---------------- K3: one LSTM step (launched T times; WMMA bf16 + TDM) ----------------
// Grid: 256 blocks of 1 wave. block = (mtile 0..7) x (h-slice 0..31).
// TDM pulls this block's W_hh slab (4 gates x 16 rows x 512 K = 64 KB bf16) into LDS as a
// 3-D tile while the wave runs the x-part GEMM from L2; s_wait_tensorcnt 0, then the
// recurrent h-part reads its B-fragments from LDS (ds_load_b128).
__global__ __launch_bounds__(32) void lstm_step(
    const float* __restrict__ seq,            // [B][T][256]
    const __bf16* __restrict__ hs_prev_base,  // hs (bf16), slot stride B*H
    __bf16* __restrict__ hs_all,              // same buffer, writable
    float* __restrict__ cfrag,                // c in fragment layout
    const float* __restrict__ Zcf,            // Zc in fragment layout
    const __bf16* __restrict__ Wih,           // [G4][256] bf16
    const __bf16* __restrict__ Whh,           // [G4][512] bf16
    int t)                                    // slot 1..T
{
    const int wg   = blockIdx.x;
    const int m0   = (wg & 7) * 16;
    const int hc0  = (wg >> 3) * 16;
    const int lane = threadIdx.x;
    const int n    = lane & 15;
    const int half = lane >> 4;

#if HAVE_TDM
    __shared__ __bf16 sWhh[4][16][H_];        // exactly 64 KB
    {
        // D# group 0: count=1, lds_addr, 57-bit global addr, type=2 ("image")
        unsigned long long ga = (unsigned long long)(const void*)(Whh + (size_t)hc0 * H_);
        u32x4 g0 = { 1u,
                     (unsigned)(size_t)(&sWhh[0][0][0]),
                     (unsigned)(ga & 0xFFFFFFFFu),
                     (unsigned)((ga >> 32) & 0x1FFFFFFu) | 0x80000000u };
        // D# group 1: data_size=2B; tensor_dim0=512, tensor_dim1=2048;
        // tile (x=512, y=16 rows, z=4 gate blocks); dim0_stride=512; dim1_stride=512*512
        i32x8 g1 = { 0x00010000,
                     (int)(512u << 16),                // tensor_dim0 low16 @ [31:16]
                     (int)(2048u << 16),               // tensor_dim1 low16 @ [31:16]
                     (int)(512u << 16),                // tile_dim0 @ [31:16]
                     (int)(16u | (4u << 16)),          // tile_dim1=16, tile_dim2=4
                     512,                              // tensor_dim0_stride low32
                     0,                                // stride0 hi / stride1 low16
                     4 };                              // tensor_dim1_stride = 4<<16 = 262144
        i32x4 g2 = { 4, 0, 0, 0 };                     // tensor_dim2 = 4
        i32x4 g3 = { 0, 0, 0, 0 };
#if __clang_major__ >= 23
        i32x8 g4 = { 0, 0, 0, 0, 0, 0, 0, 0 };
        __builtin_amdgcn_tensor_load_to_lds(g0, g1, g2, g3, g4, 0);
#else
        __builtin_amdgcn_tensor_load_to_lds(g0, g1, g2, g3, 0);
#endif
    }
#endif

    // C init: contiguous fragment-layout loads (2 x b128 per gate)
    FragF acc[4];
    {
        const f32x4* zp = (const f32x4*)(Zcf + ((size_t)((wg * 4) * 32 + lane) << 3));
        #pragma unroll
        for (int g = 0; g < 4; ++g) {
            acc[g].q[0] = zp[g * 64 + 0];
            acc[g].q[1] = zp[g * 64 + 1];
        }
    }

    const int arow = m0 + n;                  // A-fragment row for this lane
    const float* xrow = seq + ((size_t)arow * T_ + (size_t)(t - 1)) * KX_;
    if (t < T_) __builtin_prefetch(xrow + KX_, 0, 3);   // warm L2 for the next launch

    // ---- x-part: K = 256, fp32 -> bf16 native converts; B direct from L2 ----
    const __bf16* wih_base = Wih + (size_t)(hc0 + n) * KX_ + half * 16;
    #pragma unroll
    for (int kb = 0; kb < KX_ / 32; ++kb) {
        FragBF a;
        const f32x4* xp = (const f32x4*)(xrow + kb * 32 + half * 16);
        #pragma unroll
        for (int q = 0; q < 4; ++q) {
            f32x4 xv = xp[q];
            a.v[q * 4 + 0] = (__bf16)xv[0];
            a.v[q * 4 + 1] = (__bf16)xv[1];
            a.v[q * 4 + 2] = (__bf16)xv[2];
            a.v[q * 4 + 3] = (__bf16)xv[3];
        }
        FragBF bf[4];
        #pragma unroll
        for (int g = 0; g < 4; ++g) {         // batch all loads, then all WMMAs
            const __bf16* wp = wih_base + (size_t)g * (H_ * KX_) + kb * 32;
            bf[g].q[0] = *(const u32x4*)wp;
            bf[g].q[1] = *(const u32x4*)(wp + 8);
        }
        #pragma unroll
        for (int g = 0; g < 4; ++g) acc[g].v = wmma_bf16(a.v, bf[g].v, acc[g].v);
    }

#if HAVE_TDM
    __builtin_amdgcn_s_wait_tensorcnt(0);     // W_hh slab now resident in LDS
#endif

    // ---- h-part: K = 512, bf16 history; B from LDS (TDM) or L2 (fallback) ----
    const __bf16* hrow = hs_prev_base + (size_t)(t - 1) * B_ * H_ + (size_t)arow * H_;
    #pragma unroll
    for (int kb = 0; kb < H_ / 32; ++kb) {
        FragBF a;
        const __bf16* hp = hrow + kb * 32 + half * 16;
        a.q[0] = *(const u32x4*)hp;
        a.q[1] = *(const u32x4*)(hp + 8);
        FragBF bf[4];
        #pragma unroll
        for (int g = 0; g < 4; ++g) {
#if HAVE_TDM
            const __bf16* sp = &sWhh[g][n][kb * 32 + half * 16];
            bf[g].q[0] = *(const u32x4*)sp;
            bf[g].q[1] = *(const u32x4*)(sp + 8);
#else
            const __bf16* wp = Whh + (size_t)(g * H_ + hc0 + n) * H_ + kb * 32 + half * 16;
            bf[g].q[0] = *(const u32x4*)wp;
            bf[g].q[1] = *(const u32x4*)(wp + 8);
#endif
        }
        #pragma unroll
        for (int g = 0; g < 4; ++g) acc[g].v = wmma_bf16(a.v, bf[g].v, acc[g].v);
    }

    // ---- fused pointwise gate update; c stays in fragment layout (b128 RMW) ----
    float* cp = cfrag + ((size_t)(wg * 32 + lane) << 3);
    FragF cold, cnew;
    cold.q[0] = ((const f32x4*)cp)[0];
    cold.q[1] = ((const f32x4*)cp)[1];

    __bf16* hout = hs_all + (size_t)t * B_ * H_;
    #pragma unroll
    for (int i = 0; i < 8; ++i) {
        float iv = fsig(acc[0].v[i]);
        float fv = fsig(acc[1].v[i]);
        float gv = ftanh(acc[2].v[i]);
        float ov = fsig(acc[3].v[i]);
        float cn = fv * cold.v[i] + iv * gv;
        cnew.v[i] = cn;
        float hv = ov * ftanh(cn);
        int row = m0 + i + 8 * half;
        hout[(size_t)row * H_ + hc0 + n] = (__bf16)hv;
    }
    ((f32x4*)cp)[0] = cnew.q[0];
    ((f32x4*)cp)[1] = cnew.q[1];
}

// ---------------- K4: output projection hs @ W_out^T + b_out (WMMA bf16) ----------------
__global__ __launch_bounds__(256) void out_proj(
    const __bf16* __restrict__ hs, const __bf16* __restrict__ Wout_b,
    const float* __restrict__ b_out, float* __restrict__ out) {
    int wid  = (int)((blockIdx.x * blockDim.x + threadIdx.x) >> 5);
    int lane = threadIdx.x & 31;
    if (wid >= T_ * 128) return;
    int tj  = wid >> 7;
    int rem = wid & 127;
    int m0  = (rem >> 4) * 16;
    int o0  = (rem & 15) * 16;
    int n = lane & 15, half = lane >> 4;

    v8f acc;
    float bias = b_out[o0 + n];
    #pragma unroll
    for (int i = 0; i < 8; ++i) acc[i] = bias;

    const __bf16* hrow = hs + (size_t)(tj + 1) * B_ * H_ + (size_t)(m0 + n) * H_ + half * 16;
    const __bf16* wrow = Wout_b + (size_t)(o0 + n) * H_ + half * 16;
    #pragma unroll
    for (int kb = 0; kb < H_ / 32; ++kb) {
        FragBF a, bf;
        const __bf16* hp = hrow + kb * 32;
        a.q[0] = *(const u32x4*)hp;
        a.q[1] = *(const u32x4*)(hp + 8);
        const __bf16* wp = wrow + kb * 32;
        bf.q[0] = *(const u32x4*)wp;
        bf.q[1] = *(const u32x4*)(wp + 8);
        acc = wmma_bf16(a.v, bf.v, acc);
    }
    #pragma unroll
    for (int i = 0; i < 8; ++i) {
        int row = m0 + i + 8 * half;
        out[((size_t)row * T_ + tj) * OUT_ + o0 + n] = acc[i];
    }
}

// ---------------- K5a: sampled_output[:, :T-1] = seq[:, 1:], sampled_frac = 0 ----------------
__global__ void sampled_copy(const float* __restrict__ seq, float* __restrict__ out_s,
                             float* __restrict__ frac_ptr) {
    size_t tid = (size_t)blockIdx.x * blockDim.x + threadIdx.x;
    if (tid == 0) frac_ptr[0] = 0.0f;
    if (tid >= (size_t)B_ * T_ * OUT_) return;
    size_t f  = tid % OUT_;
    size_t bt = tid / OUT_;
    size_t t  = bt % T_;
    size_t b  = bt / T_;
    if (t < T_ - 1) out_s[tid] = seq[(b * T_ + (t + 1)) * OUT_ + f];
}

// ---------------- K5b: deterministic Gumbel-argmax sample of last logits ----------------
__global__ __launch_bounds__(256) void sample_last(const float* __restrict__ seq_out,
                                                   float* __restrict__ out_s) {
    __shared__ float sv[256];
    __shared__ int   si[256];
    int b = blockIdx.x, f = threadIdx.x;
    float logit = seq_out[((size_t)b * T_ + (T_ - 1)) * OUT_ + f];
    unsigned h = (unsigned)(b * OUT_ + f) * 0x9E3779B9u;
    h ^= h >> 16; h *= 0x85EBCA6Bu; h ^= h >> 13; h *= 0xC2B2AE35u; h ^= h >> 16;
    float u = ((float)(h >> 8) + 0.5f) * (1.0f / 16777216.0f);
    float g = -__logf(-__logf(u));
    sv[f] = logit + g;  // TEMPERATURE = 1.0
    si[f] = f;
    __syncthreads();
    for (int s = 128; s > 0; s >>= 1) {
        if (f < s && sv[f + s] > sv[f]) { sv[f] = sv[f + s]; si[f] = si[f + s]; }
        __syncthreads();
    }
    int am = si[0];
    out_s[((size_t)b * T_ + (T_ - 1)) * OUT_ + f] = (f == am) ? 1.0f : 0.0f;
}

// ---------------- host-side orchestration (graph-capture safe) ----------------
extern "C" void kernel_launch(void* const* d_in, const int* in_sizes, int n_in,
                              void* d_out, int out_size, void* d_ws, size_t ws_size,
                              hipStream_t stream) {
    const float* z    = (const float*)d_in[0];
    const float* seq  = (const float*)d_in[1];
    const float* We   = (const float*)d_in[2];
    const float* be   = (const float*)d_in[3];
    const float* Wih  = (const float*)d_in[4];
    const float* Whh  = (const float*)d_in[5];
    const float* bih  = (const float*)d_in[6];
    const float* bhh  = (const float*)d_in[7];
    const float* Wout = (const float*)d_in[8];
    const float* bout = (const float*)d_in[9];

    char* ws = (char*)d_ws;
    __bf16* hs    = (__bf16*)(ws + HS_OFF);
    float*  cbuf  = (float*)(ws + C_OFF);
    float*  Zcf   = (float*)(ws + ZC_OFF);
    __bf16* WihB  = (__bf16*)(ws + WIH_OFF);
    __bf16* WhhB  = (__bf16*)(ws + WHH_OFF);
    __bf16* WoutB = (__bf16*)(ws + WOUT_OFF);

    float* seq_out = (float*)d_out;                       // [B][T][256]
    float* frac    = seq_out + (size_t)B_ * T_ * OUT_;    // scalar
    float* samp    = frac + 1;                            // [B][T][256]

    embed_init<<<(B_ * H_ + 255) / 256, 256, 0, stream>>>(z, We, be, hs, cbuf);
    {
        int total = G4_ * KX_ + G4_ * H_ + OUT_ * H_;
        prep_weights<<<(total + 255) / 256, 256, 0, stream>>>(Wih, Whh, Wout, WihB, WhhB, WoutB);
    }
    zc_kernel<<<(B_ * G4_ + 255) / 256, 256, 0, stream>>>(z, Wih, bih, bhh, Zcf);

    // Sequential scan: T small-kernel launches inside the captured graph.
    for (int t = 1; t <= T_; ++t)
        lstm_step<<<NWG_, 32, 0, stream>>>(seq, hs, hs, cbuf, Zcf, WihB, WhhB, t);

    out_proj<<<(T_ * 128) / 8, 256, 0, stream>>>(hs, WoutB, bout, seq_out);

    {
        size_t total = (size_t)B_ * T_ * OUT_;
        sampled_copy<<<(int)((total + 255) / 256), 256, 0, stream>>>(seq, samp, frac);
    }
    sample_last<<<B_, 256, 0, stream>>>(seq_out, samp);
}
